// LocalDisparityRefinement_7241314861163
// MI455X (gfx1250) — matrix operand order repaired and checked by reading, not compiled, not run on hardware
//
#include <hip/hip_runtime.h>
#include <hip/hip_bf16.h>
#include <math.h>

typedef __attribute__((ext_vector_type(4)))  unsigned int u32x4;
typedef __attribute__((ext_vector_type(16))) __bf16       bf16x16;
typedef __attribute__((ext_vector_type(8)))  float        f32x8;

struct V32 { u32x4 lo, hi; };

__device__ __forceinline__ unsigned short f2bf(float f) {
  unsigned int u = __builtin_bit_cast(unsigned int, f);
  u += 0x7FFFu + ((u >> 16) & 1u);               // round-to-nearest-even
  return (unsigned short)(u >> 16);
}
__device__ __forceinline__ float bf2f(unsigned short b) {
  return __builtin_bit_cast(float, (unsigned int)b << 16);
}

// ---------------------------------------------------------------------------
// Conv inner loop. INTERIOR=true: all 9 taps in-bounds -> unconditional loads,
// no per-tap operand zeroing (half==1 zero half of K-chunk1 hoisted out of the
// loop). INTERIOR=false: per-tap, per-lane boundary guards with zero fill.
// A operands for all CT cout-tiles are preloaded per tap (one LDS clause),
// then the 4*CT WMMA chain runs.
// ---------------------------------------------------------------------------
template <int CT, bool INTERIOR>
__device__ __forceinline__ void conv_core(const unsigned short* __restrict__ src,
                                          const unsigned short* __restrict__ arow,
                                          f32x8 (&acc)[CT][2],
                                          int b, int y, int x0, int laneN, int half,
                                          int H, int W, int dil)
{
  V32 bv1[2];
  bv1[0] = {}; bv1[1] = {};        // lanes 16..31 of K-chunk1 stay zero (cin 48..63 pad)

#pragma unroll
  for (int tap = 0; tap < 9; ++tap) {
    const int dy = tap / 3 - 1, dx = tap % 3 - 1;
    int yy = y + dy * dil;
    size_t rowB = (size_t)(b * H + yy) * W;

    V32 bv0[2];
    if (INTERIOR) {
#pragma unroll
      for (int s = 0; s < 2; ++s) {
        const unsigned short* bp = src + (rowB + x0 + s * 16 + laneN + dx * dil) * 48;
        bv0[s].lo = *(const u32x4*)(bp + 16 * half);     // cin 16*half .. +7
        bv0[s].hi = *(const u32x4*)(bp + 16 * half + 8); //          .. +15
        if (half == 0) {                                 // cin 32..47
          bv1[s].lo = *(const u32x4*)(bp + 32);
          bv1[s].hi = *(const u32x4*)(bp + 40);
        }
      }
    } else {
      bool vy = (yy >= 0) && (yy < H);
#pragma unroll
      for (int s = 0; s < 2; ++s) {
        bv0[s] = {}; bv1[s] = {};
        int xx = x0 + s * 16 + laneN + dx * dil;
        if (vy && xx >= 0 && xx < W) {
          const unsigned short* bp = src + (rowB + xx) * 48;
          bv0[s].lo = *(const u32x4*)(bp + 16 * half);
          bv0[s].hi = *(const u32x4*)(bp + 16 * half + 8);
          if (half == 0) {
            bv1[s].lo = *(const u32x4*)(bp + 32);
            bv1[s].hi = *(const u32x4*)(bp + 40);
          }
        }
      }
    }
    bf16x16 Bm0[2] = { __builtin_bit_cast(bf16x16, bv0[0]), __builtin_bit_cast(bf16x16, bv0[1]) };
    bf16x16 Bm1[2] = { __builtin_bit_cast(bf16x16, bv1[0]), __builtin_bit_cast(bf16x16, bv1[1]) };

    // preload A for all cout tiles (single LDS clause), then the WMMA chain
    bf16x16 A0[CT], A1[CT];
#pragma unroll
    for (int t = 0; t < CT; ++t) {
      const unsigned short* ap = arow + (size_t)t * 16 * 576 + tap * 64;
      V32 av0, av1;
      av0.lo = *(const u32x4*)(ap);                 // K 0..7   (or 16..23)
      av0.hi = *(const u32x4*)(ap + 8);             // K 8..15  (or 24..31)
      av1.lo = *(const u32x4*)(ap + 32);            // K 32..39 (or 48..55 == 0 pad)
      av1.hi = *(const u32x4*)(ap + 40);            // K 40..47 (or 56..63 == 0 pad)
      A0[t] = __builtin_bit_cast(bf16x16, av0);
      A1[t] = __builtin_bit_cast(bf16x16, av1);
    }
#pragma unroll
    for (int t = 0; t < CT; ++t) {
#pragma unroll
      for (int s = 0; s < 2; ++s) {
        acc[t][s] = __builtin_amdgcn_wmma_f32_16x16x32_bf16(false, A0[t], false, Bm0[s], (short)0, acc[t][s], false, false);
        acc[t][s] = __builtin_amdgcn_wmma_f32_16x16x32_bf16(false, A1[t], false, Bm1[s], (short)0, acc[t][s], false, false);
      }
    }
  }
}

// ---------------------------------------------------------------------------
// Implicit-GEMM 3x3 conv, Cin=48 (K padded to 64/tap), bf16 WMMA, f32 accum.
// Weights staged in LDS once per 8-wave block (CT*16 rows x 576 K x bf16).
// One wave = 32 pixels (2 N-subtiles) x CT*16 couts. Wave-uniform interior /
// border path split. Also accumulates per-channel BN stats via atomics.
// ---------------------------------------------------------------------------
template <int CT>
__global__ __launch_bounds__(256)
void conv48_wmma(const unsigned short* __restrict__ src,   // [B,H,W,48] bf16
                 const unsigned short* __restrict__ wp,    // [CT*16][9*64] bf16
                 unsigned short* __restrict__ dst,         // [B,H,W,coutStore] bf16 (pre-BN)
                 const unsigned short* __restrict__ res,   // optional residual [B,H,W,48]
                 const float* __restrict__ bias,           // optional [Cout]
                 float* __restrict__ stats,                // [128]: sum[64], sumsq[64]
                 int H, int W, int nPixTiles, int dil, int coutStore)
{
  __shared__ __align__(16) unsigned short smem[CT * 16 * 576];
  {
    const u32x4* wsrc = (const u32x4*)wp;
    u32x4* wdst = (u32x4*)smem;
    for (int i = threadIdx.x; i < CT * 1152; i += 256) wdst[i] = wsrc[i];
  }
  __syncthreads();

  int wid = blockIdx.x * 8 + (threadIdx.x >> 5);
  if (wid >= nPixTiles) return;
  int lane  = threadIdx.x & 31;
  int laneN = lane & 15;
  int half  = lane >> 4;

  int base = wid << 5;                    // 32 pixels per wave
  int hw = H * W;
  int b   = base / hw;
  int rem = base - b * hw;
  int y   = rem / W;
  int x0  = rem - y * W;

  const unsigned short* arow = smem + (size_t)laneN * 576 + half * 16;

  f32x8 acc[CT][2];
#pragma unroll
  for (int t = 0; t < CT; ++t) {
    acc[t][0] = {};
    acc[t][1] = {};
  }

  // wave-uniform branch: all taps of all 32 pixels in-bounds?
  bool interior = (y >= dil) && (y + dil < H) && (x0 >= dil) && (x0 + 32 + dil <= W);
  if (interior)
    conv_core<CT, true>(src, arow, acc, b, y, x0, laneN, half, H, W, dil);
  else
    conv_core<CT, false>(src, arow, acc, b, y, x0, laneN, half, H, W, dil);

  size_t rowOff = (size_t)(b * H + y) * W;

#pragma unroll
  for (int t = 0; t < CT; ++t) {
    int chbase = t * 16 + half * 8;
#pragma unroll
    for (int s = 0; s < 2; ++s) {
      size_t pixOff = rowOff + x0 + s * 16 + laneN;
      if (bias) {
#pragma unroll
        for (int r = 0; r < 8; ++r) acc[t][s][r] += bias[chbase + r];
      }
      if (res) {
        union { u32x4 v; unsigned short us[8]; } ru;
        ru.v = *(const u32x4*)(res + pixOff * 48 + chbase);
#pragma unroll
        for (int r = 0; r < 8; ++r) acc[t][s][r] += bf2f(ru.us[r]);
      }
      u32x4 outv;
#pragma unroll
      for (int i = 0; i < 4; ++i)
        outv[i] = (unsigned int)f2bf(acc[t][s][2 * i]) | ((unsigned int)f2bf(acc[t][s][2 * i + 1]) << 16);
      *(u32x4*)(dst + pixOff * coutStore + chbase) = outv;
    }

    // BN statistics: combine both subtiles, butterfly the 16 pixels per half,
    // then 2 atomics per half-wave
#pragma unroll
    for (int r = 0; r < 8; ++r) {
      float sv = acc[t][0][r] + acc[t][1][r];
      float qv = acc[t][0][r] * acc[t][0][r] + acc[t][1][r] * acc[t][1][r];
#pragma unroll
      for (int m = 1; m < 16; m <<= 1) {
        sv += __shfl_xor(sv, m);
        qv += __shfl_xor(qv, m);
      }
      if (laneN == 0) {
        atomicAdd(&stats[chbase + r], sv);
        atomicAdd(&stats[64 + chbase + r], qv);
      }
    }
  }
}

// BN (batch stats) + LeakyReLU applied in place on a bf16 NHWC tensor
__global__ void bn_apply(unsigned short* __restrict__ act,
                         const float* __restrict__ stats,
                         const float* __restrict__ gamma,
                         const float* __restrict__ beta,
                         int nPix, int stride, int chCnt, int cReal, float invN)
{
  long long idx = (long long)blockIdx.x * blockDim.x + threadIdx.x;
  long long total = (long long)nPix * chCnt;
  if (idx >= total) return;
  int ch = (int)(idx % chCnt);
  long long pix = idx / chCnt;
  size_t pos = (size_t)pix * stride + ch;
  if (ch >= cReal) { act[pos] = 0; return; }
  float mean = stats[ch] * invN;
  float var  = stats[64 + ch] * invN - mean * mean;
  float sc   = gamma[ch] * rsqrtf(var + 1e-5f);
  float v    = (bf2f(act[pos]) - mean) * sc + beta[ch];
  act[pos]   = f2bf(v >= 0.f ? v : 0.01f * v);
}

// OIHW f32 -> [CoutPad][tap*64+cin] bf16, zero padded
__global__ void pack_w(const float* __restrict__ w, unsigned short* __restrict__ wp,
                       int Cout, int Cin, int CoutPad)
{
  int idx = blockIdx.x * blockDim.x + threadIdx.x;
  int total = CoutPad * 576;
  if (idx >= total) return;
  int co = idx / 576;
  int k  = idx - co * 576;
  int tap = k >> 6;
  int ci  = k & 63;
  float v = 0.f;
  if (co < Cout && ci < Cin) v = w[((size_t)co * Cin + ci) * 9 + tap];
  wp[idx] = f2bf(v);
}

__global__ void zero_stats(float* s) { if (threadIdx.x < 128) s[threadIdx.x] = 0.f; }

// ns branch conv1: 3x3, 1 -> 16 channels, on disparity; bf16 pre-BN out + stats
__global__ __launch_bounds__(256)
void ns_conv1(const float* __restrict__ disp, const float* __restrict__ w,
              unsigned short* __restrict__ out, float* __restrict__ stats,
              int H, int W, int nPix)
{
  __shared__ float ssum[16], ssq[16];
  if (threadIdx.x < 16) { ssum[threadIdx.x] = 0.f; ssq[threadIdx.x] = 0.f; }
  __syncthreads();
  int idx = blockIdx.x * 256 + threadIdx.x;
  if (idx < nPix) {
    int hw = H * W;
    int b = idx / hw, rem = idx - b * hw;
    int y = rem / W, x = rem - y * W;
    float d[9];
#pragma unroll
    for (int tap = 0; tap < 9; ++tap) {
      int yy = y + tap / 3 - 1, xx = x + tap % 3 - 1;
      d[tap] = (yy >= 0 && yy < H && xx >= 0 && xx < W) ? disp[(size_t)(b * H + yy) * W + xx] : 0.f;
    }
    float o[16];
#pragma unroll
    for (int c = 0; c < 16; ++c) {
      float s = 0.f;
#pragma unroll
      for (int tap = 0; tap < 9; ++tap) s += w[c * 9 + tap] * d[tap];
      o[c] = s;
      atomicAdd(&ssum[c], s);
      atomicAdd(&ssq[c], s * s);
    }
    u32x4 lo, hi;
#pragma unroll
    for (int i = 0; i < 4; ++i) {
      lo[i] = (unsigned int)f2bf(o[2 * i])     | ((unsigned int)f2bf(o[2 * i + 1]) << 16);
      hi[i] = (unsigned int)f2bf(o[8 + 2 * i]) | ((unsigned int)f2bf(o[9 + 2 * i]) << 16);
    }
    *(u32x4*)(out + (size_t)idx * 16)     = lo;
    *(u32x4*)(out + (size_t)idx * 16 + 8) = hi;
  }
  __syncthreads();
  if (threadIdx.x < 16) {
    atomicAdd(&stats[threadIdx.x],      ssum[threadIdx.x]);
    atomicAdd(&stats[64 + threadIdx.x], ssq[threadIdx.x]);
  }
}

// ns branch conv2: 1x1, 16 -> 8, writes into x0 channels 0..7
__global__ __launch_bounds__(256)
void ns_conv2(const unsigned short* __restrict__ in, const float* __restrict__ w,
              unsigned short* __restrict__ x0, float* __restrict__ stats, int nPix)
{
  __shared__ float ssum[8], ssq[8];
  if (threadIdx.x < 8) { ssum[threadIdx.x] = 0.f; ssq[threadIdx.x] = 0.f; }
  __syncthreads();
  int idx = blockIdx.x * 256 + threadIdx.x;
  if (idx < nPix) {
    union { u32x4 v; unsigned short s[8]; } a0, a1;
    a0.v = *(const u32x4*)(in + (size_t)idx * 16);
    a1.v = *(const u32x4*)(in + (size_t)idx * 16 + 8);
    float v[16];
#pragma unroll
    for (int c = 0; c < 8; ++c) { v[c] = bf2f(a0.s[c]); v[8 + c] = bf2f(a1.s[c]); }
    u32x4 outv;
#pragma unroll
    for (int o = 0; o < 8; ++o) {
      float s = 0.f;
#pragma unroll
      for (int c = 0; c < 16; ++c) s += w[o * 16 + c] * v[c];
      atomicAdd(&ssum[o], s);
      atomicAdd(&ssq[o], s * s);
      if (o & 1) outv[o >> 1] |= (unsigned int)f2bf(s) << 16;
      else       outv[o >> 1]  = (unsigned int)f2bf(s);
    }
    *(u32x4*)(x0 + (size_t)idx * 48) = outv;
  }
  __syncthreads();
  if (threadIdx.x < 8) {
    atomicAdd(&stats[threadIdx.x],      ssum[threadIdx.x]);
    atomicAdd(&stats[64 + threadIdx.x], ssq[threadIdx.x]);
  }
}

// Warp-and-aggregate cost volume -> x0 channels 8..47
__global__ __launch_bounds__(256)
void cost_kernel(const float* __restrict__ disp, const float* __restrict__ left,
                 const float* __restrict__ right, unsigned short* __restrict__ x0,
                 int H, int W, int nPix)
{
  int idx = blockIdx.x * 256 + threadIdx.x;
  if (idx >= nPix) return;
  int hw = H * W;
  int b = idx / hw, rem = idx - b * hw;
  int y = rem / W, x = rem - y * W;
  size_t chS = (size_t)hw;
  size_t base0 = (size_t)b * 32 * chS + (size_t)y * W + x;

  float l[32];
#pragma unroll
  for (int c = 0; c < 32; ++c) l[c] = left[base0 + c * chS];
  float ln[8];
#pragma unroll
  for (int g = 0; g < 8; ++g) {
    float s = 0.f;
#pragma unroll
    for (int j = 0; j < 4; ++j) s += l[g * 4 + j] * l[g * 4 + j];
    ln[g] = 1.f / (sqrtf(s) + 1e-5f);
  }
  float dr = (float)x - disp[idx];
  const float offs[5] = {2.f, 1.f, 0.f, -1.f, -2.f};
  size_t rowBase = base0 - (size_t)x;   // right[b, c, y, :] base (add c*chS + col)
  for (int o = 0; o < 5; ++o) {
    float idx_f = dr + offs[o];
    float fl = floorf(idx_f);
    float il = fminf(fmaxf(fl, 0.f), (float)(W - 1));
    float ir = fminf(fmaxf(fl + 1.f, 0.f), (float)(W - 1));
    float wgt = idx_f - il;
    int iil = (int)il, iir = (int)ir;
#pragma unroll
    for (int g = 0; g < 8; ++g) {
      float rw[4], ssq = 0.f;
#pragma unroll
      for (int j = 0; j < 4; ++j) {
        size_t cb = rowBase + (size_t)(g * 4 + j) * chS;
        float rl = right[cb + iil];
        float rr = right[cb + iir];
        rw[j] = rl + wgt * (rr - rl);
        ssq += rw[j] * rw[j];
      }
      float rn = 1.f / (sqrtf(ssq) + 1e-5f);
      float s = 0.f;
#pragma unroll
      for (int j = 0; j < 4; ++j) s += fabsf(l[g * 4 + j] * ln[g] - rw[j] * rn);
      x0[(size_t)idx * 48 + 8 + o * 8 + g] = f2bf(s);
    }
  }
}

// cv conv (3x3, 12->5, +bias) + softmax + expectation + (disp+err)/2
__global__ __launch_bounds__(256)
void final_kernel(const unsigned short* __restrict__ c1act, const float* __restrict__ cvw,
                  const float* __restrict__ cvb, const float* __restrict__ disp,
                  float* __restrict__ out, int H, int W, int nPix)
{
  int idx = blockIdx.x * 256 + threadIdx.x;
  if (idx >= nPix) return;
  int hw = H * W;
  int b = idx / hw, rem = idx - b * hw;
  int y = rem / W, x = rem - y * W;
  float logit[5];
#pragma unroll
  for (int o = 0; o < 5; ++o) logit[o] = cvb[o];
  for (int tap = 0; tap < 9; ++tap) {
    int yy = y + tap / 3 - 1, xx = x + tap % 3 - 1;
    if (yy < 0 || yy >= H || xx < 0 || xx >= W) continue;
    const unsigned short* p = c1act + ((size_t)(b * H + yy) * W + xx) * 16;
#pragma unroll
    for (int c = 0; c < 12; ++c) {
      float a = bf2f(p[c]);
#pragma unroll
      for (int o = 0; o < 5; ++o) logit[o] += cvw[((size_t)o * 12 + c) * 9 + tap] * a;
    }
  }
  float mx = logit[0];
#pragma unroll
  for (int o = 1; o < 5; ++o) mx = fmaxf(mx, logit[o]);
  float e[5], sum = 0.f;
#pragma unroll
  for (int o = 0; o < 5; ++o) { e[o] = expf(logit[o] - mx); sum += e[o]; }
  const float bias_list[5] = {2.f, 1.f, 0.f, -1.f, -2.f};
  float err = 0.f;
#pragma unroll
  for (int o = 0; o < 5; ++o) err += (e[o] / sum) * bias_list[o];
  out[idx] = (disp[idx] + err) * 0.5f;
}

// ---------------------------------------------------------------------------
extern "C" void kernel_launch(void* const* d_in, const int* in_sizes, int n_in,
                              void* d_out, int out_size, void* d_ws, size_t ws_size,
                              hipStream_t stream)
{
  (void)in_sizes; (void)n_in; (void)out_size; (void)ws_size;
  const float* disp  = (const float*)d_in[0];
  const float* left  = (const float*)d_in[1];
  const float* right = (const float*)d_in[2];
  const float* ns_w1 = (const float*)d_in[3];
  const float* ns_g1 = (const float*)d_in[4];
  const float* ns_b1 = (const float*)d_in[5];
  const float* ns_w2 = (const float*)d_in[6];
  const float* ns_g2 = (const float*)d_in[7];
  const float* ns_b2 = (const float*)d_in[8];
  const float* cn_w  = (const float*)d_in[9];
  const float* cn_g  = (const float*)d_in[10];
  const float* cn_b  = (const float*)d_in[11];
  auto rbp = [&](int i, int j) { return (const float*)d_in[12 + i * 7 + j]; };
  const float* c1_w = (const float*)d_in[40];
  const float* c1_g = (const float*)d_in[41];
  const float* c1_b = (const float*)d_in[42];
  const float* cv_w = (const float*)d_in[43];
  const float* cv_b = (const float*)d_in[44];

  constexpr int Bn = 4, H = 256, Wd = 512;
  constexpr size_t NPIX = (size_t)Bn * H * Wd;   // 524288

  char* ws = (char*)d_ws;
  size_t off = 0;
  auto wsa = [&](size_t bytes) { void* p = ws + off; off = (off + bytes + 255) & ~(size_t)255; return p; };
  unsigned short* actA = (unsigned short*)wsa(NPIX * 48 * 2);
  unsigned short* actB = (unsigned short*)wsa(NPIX * 48 * 2);
  unsigned short* actC = (unsigned short*)wsa(NPIX * 48 * 2);
  unsigned short* ns1t = (unsigned short*)wsa(NPIX * 16 * 2);
  unsigned short* c1o  = (unsigned short*)wsa(NPIX * 16 * 2);
  unsigned short* wpCn = (unsigned short*)wsa(48 * 576 * 2);
  unsigned short* wpRb[8];
  for (int i = 0; i < 8; ++i) wpRb[i] = (unsigned short*)wsa(48 * 576 * 2);
  unsigned short* wpC1 = (unsigned short*)wsa(16 * 576 * 2);
  float* stats = (float*)wsa(128 * 4);

  const float invN = 1.f / (float)NPIX;
  dim3 blk(256);
  int nsBlocks = (int)((NPIX + 255) / 256);

  // weight packing
  pack_w<<<(48 * 576 + 255) / 256, blk, 0, stream>>>(cn_w, wpCn, 48, 48, 48);
  for (int i = 0; i < 4; ++i) {
    pack_w<<<(48 * 576 + 255) / 256, blk, 0, stream>>>(rbp(i, 0), wpRb[2 * i], 48, 48, 48);
    pack_w<<<(48 * 576 + 255) / 256, blk, 0, stream>>>(rbp(i, 3), wpRb[2 * i + 1], 48, 48, 48);
  }
  pack_w<<<(16 * 576 + 255) / 256, blk, 0, stream>>>(c1_w, wpC1, 12, 48, 16);

  // ns branch
  zero_stats<<<1, 128, 0, stream>>>(stats);
  ns_conv1<<<nsBlocks, blk, 0, stream>>>(disp, ns_w1, ns1t, stats, H, Wd, (int)NPIX);
  bn_apply<<<(int)((NPIX * 16 + 255) / 256), blk, 0, stream>>>(ns1t, stats, ns_g1, ns_b1, (int)NPIX, 16, 16, 16, invN);
  zero_stats<<<1, 128, 0, stream>>>(stats);
  ns_conv2<<<nsBlocks, blk, 0, stream>>>(ns1t, ns_w2, actA, stats, (int)NPIX);
  bn_apply<<<(int)((NPIX * 8 + 255) / 256), blk, 0, stream>>>(actA, stats, ns_g2, ns_b2, (int)NPIX, 48, 8, 8, invN);

  // cost volume -> x0 channels 8..47 (x0 == actA)
  cost_kernel<<<nsBlocks, blk, 0, stream>>>(disp, left, right, actA, H, Wd, (int)NPIX);

  int pixTiles = (int)(NPIX / 32);          // 32 pixels per wave
  int convBlocks = (pixTiles + 7) / 8;

  auto convbn48 = [&](const unsigned short* src, const unsigned short* wp, unsigned short* dst,
                      const unsigned short* res, const float* bias, const float* g, const float* be,
                      int dil) {
    zero_stats<<<1, 128, 0, stream>>>(stats);
    conv48_wmma<3><<<convBlocks, blk, 0, stream>>>(src, wp, dst, res, bias, stats,
                                                   H, Wd, pixTiles, dil, 48);
    bn_apply<<<(int)((NPIX * 48 + 255) / 256), blk, 0, stream>>>(
        dst, stats, g, be, (int)NPIX, 48, 48, 48, invN);
  };

  // cn: x0(A) -> B
  convbn48(actA, wpCn, actB, nullptr, nullptr, cn_g, cn_b, 1);

  const int DIL[4] = {1, 2, 1, 1};
  unsigned short* bufs[3] = {actA, actB, actC};
  int xi = 1;  // current x = actB
  for (int i = 0; i < 4; ++i) {
    int ti = (xi + 1) % 3, oi = (xi + 2) % 3;
    convbn48(bufs[xi], wpRb[2 * i], bufs[ti], nullptr, nullptr, rbp(i, 1), rbp(i, 2), DIL[i]);
    convbn48(bufs[ti], wpRb[2 * i + 1], bufs[oi], bufs[xi], rbp(i, 4), rbp(i, 5), rbp(i, 6), DIL[i]);
    xi = oi;
  }

  // c1: x -> c1o (12 real couts padded to 16)
  zero_stats<<<1, 128, 0, stream>>>(stats);
  conv48_wmma<1><<<convBlocks, blk, 0, stream>>>(bufs[xi], wpC1, c1o, nullptr, nullptr, stats,
                                                 H, Wd, pixTiles, 1, 16);
  bn_apply<<<(int)((NPIX * 16 + 255) / 256), blk, 0, stream>>>(
      c1o, stats, c1_g, c1_b, (int)NPIX, 16, 16, 12, invN);

  // cv + softmax + expectation + blend
  final_kernel<<<nsBlocks, blk, 0, stream>>>(c1o, cv_w, cv_b, disp, (float*)d_out, H, Wd, (int)NPIX);
}
